// Transformer_61220463837536
// MI455X (gfx1250) — matrix operand order repaired and checked by reading, not compiled, not run on hardware
//
#include <hip/hip_runtime.h>
#include <hip/hip_bf16.h>

// Problem constants (match reference)
#define LNUM   6
#define VOC    256
#define DMODEL 512
#define NH     8
#define SEQ    1024
#define BATCH  4
#define DKH    64
#define FFD    2048
#define NROWS  (BATCH*SEQ)   // 4096

typedef __bf16 bf16_t;
typedef __attribute__((ext_vector_type(16))) __bf16        v16bf;
typedef __attribute__((ext_vector_type(8)))  float         v8f;
typedef __attribute__((ext_vector_type(4)))  unsigned int  u32x4;
typedef __attribute__((ext_vector_type(4)))  unsigned int  uint32x4;
typedef __attribute__((ext_vector_type(4)))  int           int32x4;
typedef __attribute__((ext_vector_type(8)))  int           int32x8;

#if __has_builtin(__builtin_amdgcn_tensor_load_to_lds) && __has_builtin(__builtin_amdgcn_s_wait_tensorcnt)
#define USE_TDM 1
#else
#define USE_TDM 0
#endif

union FragU { u32x4 q[2]; v16bf v; };   // 32B: one WMMA bf16 A/B fragment

static __device__ __forceinline__ v8f wmma_bf16(const FragU& a, const FragU& b, v8f c) {
  return __builtin_amdgcn_wmma_f32_16x16x32_bf16(false, a.v, false, b.v,
                                                 (short)0, c, false, false);
}

#if USE_TDM
// 32-bit LDS byte address of a __shared__ object (generic -> AS3 -> int)
static __device__ __forceinline__ unsigned lds_addr_of(const void* p) {
  return (unsigned)(unsigned long long)(__attribute__((address_space(3))) const void*)p;
}

// Issue one 2D-tile TDM load: tileH rows x tileW bf16 elems, row stride strideElems,
// LDS rows padded: padIntervalCode -> 2^(c+1) DWORDs between pads, padAmountCode -> c+1 DWORDs pad.
static __device__ __forceinline__ void tdm_load_2d(const bf16_t* gsrc, unsigned ldsByte,
                                                   unsigned tensorW, unsigned strideElems,
                                                   unsigned tileW, unsigned tileH,
                                                   unsigned padIntervalCode, unsigned padAmountCode) {
  unsigned long long ga = (unsigned long long)(size_t)gsrc;
  uint32x4 g0;
  g0.x = 1u;                                               // count=1 (valid descriptor)
  g0.y = ldsByte;                                          // lds_addr
  g0.z = (unsigned)(ga & 0xFFFFFFFFu);                     // global_addr[31:0]
  g0.w = (unsigned)((ga >> 32) & 0x01FFFFFFu) | (2u << 30);// global_addr[56:32] | type=2
  int32x8 g1;
  const unsigned flags = (1u << 16)                        // data_size = 2 bytes
                       | (1u << 20)                        // pad_enable
                       | (padIntervalCode << 22)
                       | (padAmountCode << 25);
  g1[0] = (int)flags;                                      // wg_mask=0, no atomic barrier
  g1[1] = (int)((tensorW & 0xFFFFu) << 16);                // tensor_dim0[15:0] @ bit48
  g1[2] = (int)(tensorW >> 16);                            // tensor_dim0[31:16] | tensor_dim1[15:0]=0
  g1[3] = (int)(0x10u | (tileW << 16));                    // tensor_dim1[31:16]=0x10 (1M rows) | tile_dim0
  g1[4] = (int)(tileH & 0xFFFFu);                          // tile_dim1 | tile_dim2=0 (2D)
  g1[5] = (int)strideElems;                                // tensor_dim0_stride[31:0]
  g1[6] = 0;                                               // stride[47:32]=0 | dim1_stride lo = 0
  g1[7] = 0;
  const int32x4 z4 = {0, 0, 0, 0};
#if defined(__clang_major__) && (__clang_major__ >= 23)
  const int32x8 z8 = {0, 0, 0, 0, 0, 0, 0, 0};
  __builtin_amdgcn_tensor_load_to_lds(g0, g1, z4, z4, z8, 0);
#else
  __builtin_amdgcn_tensor_load_to_lds(g0, g1, z4, z4, 0);
#endif
}
#endif // USE_TDM

// ---------------------------------------------------------------- convert
__global__ void convert_f32_bf16(const float* __restrict__ in, bf16_t* __restrict__ out, int n) {
  int i = blockIdx.x * blockDim.x + threadIdx.x;
  int stride = gridDim.x * blockDim.x;
  for (; i < n; i += stride) out[i] = (bf16_t)in[i];
}

// ---------------------------------------------------------------- layernorm (wave per row, D=512)
__global__ __launch_bounds__(256) void ln_kernel(const float* __restrict__ h,
                                                 const float* __restrict__ g,
                                                 const float* __restrict__ b,
                                                 bf16_t* __restrict__ out) {
  const int w = threadIdx.x >> 5, lane = threadIdx.x & 31;
  const int row = blockIdx.x * 8 + w;
  const float* x = h + (size_t)row * DMODEL;
  float vals[16];
  float s = 0.f, sq = 0.f;
#pragma unroll
  for (int i = 0; i < 16; ++i) { float v = x[lane + 32*i]; vals[i] = v; s += v; sq += v*v; }
#pragma unroll
  for (int m = 1; m <= 16; m <<= 1) { s += __shfl_xor(s, m); sq += __shfl_xor(sq, m); }
  const float mean = s * (1.f/DMODEL);
  const float var  = sq * (1.f/DMODEL) - mean*mean;
  const float rs   = rsqrtf(var + 1e-5f);
  bf16_t* o = out + (size_t)row * DMODEL;
#pragma unroll
  for (int i = 0; i < 16; ++i) {
    int d = lane + 32*i;
    o[d] = (bf16_t)((vals[i] - mean) * rs * g[d] + b[d]);
  }
}

// ---------------------------------------------------------------- GEMM  C[M,N] = A[M,K] @ W[K,N]
// A,W bf16 row-major.  128x128 block tile, BK=32, 8 waves -> each wave 32x64 (8 WMMA tiles).
// Tiles staged into LDS by the Tensor Data Mover (double-buffered, TENSORcnt) when available.
enum { EP_EMB = 0, EP_BF16 = 1, EP_KT = 2, EP_RESID = 3, EP_GELU = 4, EP_OUT = 5 };

template <int MODE>
__global__ void __launch_bounds__(256) gemm_bf16(
    const bf16_t* __restrict__ A, const bf16_t* __restrict__ W,
    const float* __restrict__ bias, const float* extra,
    float* outF, bf16_t* outB, int M, int N, int K) {
#if USE_TDM
  __shared__ bf16_t lA[2][128 * 40];   // row-major, stride 40 (pad: 16DW + 4DW)
  __shared__ bf16_t lB[2][32 * 136];   // row-major, stride 136 (pad: 64DW + 4DW)
#else
  __shared__ bf16_t lA[1][128 * 40];
  __shared__ bf16_t lB[1][32 * 136];
#endif
  const int tid   = threadIdx.x;
  const int wid   = tid >> 5, lane = tid & 31;
  const int n_l   = lane & 15, half = lane >> 4, base0 = half * 8;
  const int rowBase = blockIdx.y * 128;
  const int colBase = blockIdx.x * 128;
  const int wRow = (wid >> 1) * 32;   // 0,32,64,96
  const int wCol = (wid & 1) * 64;    // 0,64

  v8f acc[2][4] = {};

#if USE_TDM
  const int nTiles = K >> 5;
  if (wid == 0) {
    tdm_load_2d(A + (size_t)rowBase * K, lds_addr_of(&lA[0][0]), K, K, 32, 128, 3, 3);
    tdm_load_2d(W + colBase,             lds_addr_of(&lB[0][0]), N, N, 128, 32, 5, 3);
  }
  for (int it = 0; it < nTiles; ++it) {
    const int cur = it & 1;
    if (wid == 0) {
      if (it + 1 < nTiles) {
        const int k0n = (it + 1) << 5;
        tdm_load_2d(A + (size_t)rowBase * K + k0n, lds_addr_of(&lA[cur ^ 1][0]), K, K, 32, 128, 3, 3);
        tdm_load_2d(W + (size_t)k0n * N + colBase, lds_addr_of(&lB[cur ^ 1][0]), N, N, 128, 32, 5, 3);
        __builtin_amdgcn_s_wait_tensorcnt(2);   // current tile's 2 DMAs complete
      } else {
        __builtin_amdgcn_s_wait_tensorcnt(0);
      }
    }
    __syncthreads();
#else
  const int aRow = tid >> 1, aCol = (tid & 1) * 16;   // A tile fill: 128 rows x 32
  const int bK   = tid >> 3, bN   = (tid & 7) * 16;   // B tile fill: 32 rows x 128
  for (int it = 0; it < (K >> 5); ++it) {
    const int cur = 0;
    const int k0 = it << 5;
    __syncthreads();
    {
      const bf16_t* src = A + (size_t)(rowBase + aRow) * K + k0 + aCol;
      *(u32x4*)&lA[0][aRow * 40 + aCol]     = *(const u32x4*)src;
      *(u32x4*)&lA[0][aRow * 40 + aCol + 8] = *(const u32x4*)(src + 8);
    }
    {
      const bf16_t* src = W + (size_t)(k0 + bK) * N + colBase + bN;
      *(u32x4*)&lB[0][bK * 136 + bN]     = *(const u32x4*)src;
      *(u32x4*)&lB[0][bK * 136 + bN + 8] = *(const u32x4*)(src + 8);
    }
    __syncthreads();
#endif
    FragU fa[2], fb[4];
#pragma unroll
    for (int mt = 0; mt < 2; ++mt) {
      const bf16_t* p = &lA[cur][(wRow + mt * 16 + n_l) * 40];
      fa[mt].q[0] = *(const u32x4*)(p + base0);
      fa[mt].q[1] = *(const u32x4*)(p + 16 + base0);
    }
#pragma unroll
    for (int nt = 0; nt < 4; ++nt) {
      const bf16_t* p = &lB[cur][wCol + nt * 16 + n_l];
#pragma unroll
      for (int e = 0; e < 16; ++e) fb[nt].v[e] = p[(e < 8 ? base0 + e : 8 + base0 + e) * 136];
    }
#pragma unroll
    for (int mt = 0; mt < 2; ++mt)
#pragma unroll
      for (int nt = 0; nt < 4; ++nt)
        acc[mt][nt] = wmma_bf16(fa[mt], fb[nt], acc[mt][nt]);
#if USE_TDM
    __syncthreads();   // protect buffer before TDM refill next iteration
#endif
  }

  // Epilogue: C layout lane holds col n_l, rows r+8*half
#pragma unroll
  for (int mt = 0; mt < 2; ++mt)
#pragma unroll
    for (int nt = 0; nt < 4; ++nt)
#pragma unroll
      for (int r = 0; r < 8; ++r) {
        const int gm = rowBase + wRow + mt * 16 + r + 8 * half;
        const int gn = colBase + wCol + nt * 16 + n_l;
        float v = acc[mt][nt][r];
        if constexpr (MODE == EP_EMB) {
          outF[(size_t)gm * N + gn] = v + bias[gn] + extra[(size_t)(gm & (SEQ - 1)) * N + gn];
        } else if constexpr (MODE == EP_BF16) {
          outB[(size_t)gm * N + gn] = (bf16_t)v;
        } else if constexpr (MODE == EP_KT) {
          const int head = gn >> 6, dk = gn & 63;
          const int bb = gm >> 10, ss = gm & (SEQ - 1);
          outB[(size_t)((bb * NH + head) * DKH + dk) * SEQ + ss] = (bf16_t)v;
        } else if constexpr (MODE == EP_RESID) {
          outF[(size_t)gm * N + gn] = extra[(size_t)gm * N + gn] + v + bias[gn];
        } else if constexpr (MODE == EP_GELU) {
          float t = v + bias[gn];
          outB[(size_t)gm * N + gn] = (bf16_t)(0.5f * t * (1.f + erff(t * 0.70710678118f)));
        } else { // EP_OUT
          outF[(size_t)gm * N + gn] = v;
        }
      }
}

// ---------------------------------------------------------------- flash attention
// One 32-key chunk: scores (2x16-key tiles, 2 WMMA each) + online softmax + P@V (4 WMMA).
template <bool MASKED>
static __device__ __forceinline__ void attn_chunk(
    int j, int q0, int n_l, int half, int base0, int lane, int b, int bh, int h,
    const FragU* fq, const bf16_t* __restrict__ kt, const bf16_t* __restrict__ vb,
    bf16_t* pb, v8f* o, float* mrow, float* lrow) {
  const float scale = 0.125f;   // 1/sqrt(DK)
  v8f sc[2];
#pragma unroll
  for (int t = 0; t < 2; ++t) {
    const int kbase = j + 16 * t;
    v8f c = {};
#pragma unroll
    for (int s = 0; s < 2; ++s) {
      // B fragment of K^T: lane = key column (kbase+n_l), 16 dk rows in table order
      const bf16_t* kp = kt + (size_t)(bh * DKH + 32 * s) * SEQ + kbase + n_l;
      FragU fk;
#pragma unroll
      for (int e = 0; e < 16; ++e) fk.v[e] = kp[(size_t)(e < 8 ? base0 + e : 8 + base0 + e) * SEQ];
      c = wmma_bf16(fq[s], fk, c);
    }
    sc[t] = c;
  }
  float tm[8], rs[8];
#pragma unroll
  for (int r = 0; r < 8; ++r) {
    float s0 = sc[0][r] * scale;
    float s1 = sc[1][r] * scale;
    if constexpr (MASKED) {
      const int gq = q0 + r + 8 * half;
      s0 = (j + n_l      <= gq) ? s0 : -1e30f;
      s1 = (j + 16 + n_l <= gq) ? s1 : -1e30f;
    }
    sc[0][r] = s0; sc[1][r] = s1;
    tm[r] = fmaxf(s0, s1);
  }
#pragma unroll
  for (int m = 1; m <= 8; m <<= 1)
#pragma unroll
    for (int r = 0; r < 8; ++r) tm[r] = fmaxf(tm[r], __shfl_xor(tm[r], m));
#pragma unroll
  for (int r = 0; r < 8; ++r) {
    const float mnew  = fmaxf(mrow[r], tm[r]);
    const float alpha = __expf(mrow[r] - mnew);
    mrow[r] = mnew;
    float p0, p1;
    if constexpr (MASKED) {
      p0 = (sc[0][r] > -1e29f) ? __expf(sc[0][r] - mnew) : 0.f;
      p1 = (sc[1][r] > -1e29f) ? __expf(sc[1][r] - mnew) : 0.f;
    } else {
      p0 = __expf(sc[0][r] - mnew);
      p1 = __expf(sc[1][r] - mnew);
    }
    pb[(r + 8 * half) * 40 + n_l]      = (bf16_t)p0;
    pb[(r + 8 * half) * 40 + 16 + n_l] = (bf16_t)p1;
    rs[r] = p0 + p1;
    lrow[r] *= alpha;
#pragma unroll
    for (int d = 0; d < 4; ++d) o[d][r] = o[d][r] * alpha;
  }
#pragma unroll
  for (int m = 1; m <= 8; m <<= 1)
#pragma unroll
    for (int r = 0; r < 8; ++r) rs[r] += __shfl_xor(rs[r], m);
#pragma unroll
  for (int r = 0; r < 8; ++r) lrow[r] += rs[r];
  // LDS re-shape of probs: C layout -> A layout (per-wave buffer, wait DS in-order)
  asm volatile("s_wait_dscnt 0" ::: "memory");
  FragU fp;
  fp.q[0] = *(const u32x4*)&pb[n_l * 40 + base0];
  fp.q[1] = *(const u32x4*)&pb[n_l * 40 + 16 + base0];
#pragma unroll
  for (int d = 0; d < 4; ++d) {
    // B fragment of V: lane = dk column (d*16+n_l), 16 key rows in table order
    const bf16_t* vp = vb + (size_t)(b * SEQ + j) * DMODEL + h * DKH + d * 16 + n_l;
    FragU fv;
#pragma unroll
    for (int e = 0; e < 16; ++e) fv.v[e] = vp[(size_t)(e < 8 ? base0 + e : 8 + base0 + e) * DMODEL];
    o[d] = wmma_bf16(fp, fv, o[d]);
  }
}

// q: [B,S,D] bf16; kt: [B,H,DK,S] bf16; v: [B,S,D] bf16; y: [B,S,D] bf16
__global__ void __launch_bounds__(256) attn_kernel(const bf16_t* __restrict__ qb,
                                                   const bf16_t* __restrict__ kt,
                                                   const bf16_t* __restrict__ vb,
                                                   bf16_t* __restrict__ yb) {
  __shared__ bf16_t pshared[8 * 16 * 40];
  const int tid  = threadIdx.x;
  const int wid  = tid >> 5, lane = tid & 31;
  const int n_l  = lane & 15, half = lane >> 4, base0 = half * 8;
  const int bh   = blockIdx.x / (SEQ / 128);
  const int qblk = blockIdx.x % (SEQ / 128);
  const int b    = bh / NH, h = bh % NH;
  const int q0   = qblk * 128 + wid * 16;
  bf16_t* pb = &pshared[wid * 16 * 40];

  // Q fragments: A layout (lane = row, dk split per half)
  FragU fq[2];
  {
    const bf16_t* qp = qb + (size_t)(b * SEQ + q0 + n_l) * DMODEL + h * DKH;
#pragma unroll
    for (int s = 0; s < 2; ++s) {
      fq[s].q[0] = *(const u32x4*)(qp + 32 * s + base0);
      fq[s].q[1] = *(const u32x4*)(qp + 32 * s + 16 + base0);
    }
  }

  v8f o[4] = {};
  float mrow[8], lrow[8];
#pragma unroll
  for (int r = 0; r < 8; ++r) { mrow[r] = -1e30f; lrow[r] = 0.f; }

  // Chunks fully below the diagonal need no mask: j+31 <= q0.
  const int jFull = (q0 >= 31) ? ((((q0 - 31) >> 5) << 5) + 32) : 0;
  int j = 0;
  for (; j < jFull; j += 32)
    attn_chunk<false>(j, q0, n_l, half, base0, lane, b, bh, h, fq, kt, vb, pb, o, mrow, lrow);
  for (; j < q0 + 16; j += 32)
    attn_chunk<true >(j, q0, n_l, half, base0, lane, b, bh, h, fq, kt, vb, pb, o, mrow, lrow);

#pragma unroll
  for (int d = 0; d < 4; ++d)
#pragma unroll
    for (int r = 0; r < 8; ++r) {
      float v = o[d][r] / lrow[r];
      yb[(size_t)(b * SEQ + q0 + r + 8 * half) * DMODEL + h * DKH + d * 16 + n_l] = (bf16_t)v;
    }
}

// ---------------------------------------------------------------- host
extern "C" void kernel_launch(void* const* d_in, const int* in_sizes, int n_in,
                              void* d_out, int out_size, void* d_ws, size_t ws_size,
                              hipStream_t stream) {
  (void)in_sizes; (void)n_in; (void)out_size; (void)ws_size;
  const float* x     = (const float*)d_in[0];
  const float* w_emb = (const float*)d_in[1];
  const float* b_emb = (const float*)d_in[2];
  const float* pos   = (const float*)d_in[3];
  const float* ln1_g = (const float*)d_in[4];
  const float* ln1_b = (const float*)d_in[5];
  const float* wq    = (const float*)d_in[6];
  const float* wk    = (const float*)d_in[7];
  const float* wv    = (const float*)d_in[8];
  const float* wp    = (const float*)d_in[9];
  const float* bp    = (const float*)d_in[10];
  const float* ln2_g = (const float*)d_in[11];
  const float* ln2_b = (const float*)d_in[12];
  const float* w1    = (const float*)d_in[13];
  const float* b1    = (const float*)d_in[14];
  const float* w2    = (const float*)d_in[15];
  const float* b2    = (const float*)d_in[16];
  const float* lnf_g = (const float*)d_in[17];
  const float* lnf_b = (const float*)d_in[18];
  const float* w_out = (const float*)d_in[19];

  char* p = (char*)d_ws;
  auto alloc = [&](size_t elems, size_t esz) -> void* {
    void* r = (void*)p;
    p += ((elems * esz + 255) & ~(size_t)255);
    return r;
  };
  bf16_t* xb    = (bf16_t*)alloc((size_t)NROWS * VOC, 2);
  bf16_t* wembB = (bf16_t*)alloc((size_t)VOC * DMODEL, 2);
  bf16_t* wqB   = (bf16_t*)alloc((size_t)LNUM * DMODEL * DMODEL, 2);
  bf16_t* wkB   = (bf16_t*)alloc((size_t)LNUM * DMODEL * DMODEL, 2);
  bf16_t* wvB   = (bf16_t*)alloc((size_t)LNUM * DMODEL * DMODEL, 2);
  bf16_t* wpB   = (bf16_t*)alloc((size_t)LNUM * DMODEL * DMODEL, 2);
  bf16_t* w1B   = (bf16_t*)alloc((size_t)LNUM * DMODEL * FFD, 2);
  bf16_t* w2B   = (bf16_t*)alloc((size_t)LNUM * FFD * DMODEL, 2);
  bf16_t* woutB = (bf16_t*)alloc((size_t)DMODEL * VOC, 2);
  float*  hbuf  = (float*) alloc((size_t)NROWS * DMODEL, 4);
  bf16_t* hn    = (bf16_t*)alloc((size_t)NROWS * DMODEL, 2);
  bf16_t* qB    = (bf16_t*)alloc((size_t)NROWS * DMODEL, 2);
  bf16_t* ktB   = (bf16_t*)alloc((size_t)NROWS * DMODEL, 2);
  bf16_t* vB    = (bf16_t*)alloc((size_t)NROWS * DMODEL, 2);
  bf16_t* yB    = (bf16_t*)alloc((size_t)NROWS * DMODEL, 2);
  bf16_t* ffB   = (bf16_t*)alloc((size_t)NROWS * FFD, 2);
  bf16_t* h2B   = (bf16_t*)alloc((size_t)NROWS * DMODEL, 2);

  auto conv = [&](const float* in, bf16_t* out, size_t n) {
    convert_f32_bf16<<<dim3(1024), dim3(256), 0, stream>>>(in, out, (int)n);
  };
  conv(x,     xb,    (size_t)NROWS * VOC);
  conv(w_emb, wembB, (size_t)VOC * DMODEL);
  conv(wq,    wqB,   (size_t)LNUM * DMODEL * DMODEL);
  conv(wk,    wkB,   (size_t)LNUM * DMODEL * DMODEL);
  conv(wv,    wvB,   (size_t)LNUM * DMODEL * DMODEL);
  conv(wp,    wpB,   (size_t)LNUM * DMODEL * DMODEL);
  conv(w1,    w1B,   (size_t)LNUM * DMODEL * FFD);
  conv(w2,    w2B,   (size_t)LNUM * FFD * DMODEL);
  conv(w_out, woutB, (size_t)DMODEL * VOC);

  const dim3 blk(256);
  const dim3 gD(DMODEL / 128, NROWS / 128);   // N=512
  const dim3 gF(FFD / 128,    NROWS / 128);   // N=2048
  const dim3 gV(VOC / 128,    NROWS / 128);   // N=256

  // h = x @ w_emb + b_emb + pos
  gemm_bf16<EP_EMB><<<gD, blk, 0, stream>>>(xb, wembB, b_emb, pos, hbuf, nullptr,
                                            NROWS, DMODEL, VOC);

  for (int l = 0; l < LNUM; ++l) {
    const size_t wOff = (size_t)l * DMODEL * DMODEL;
    ln_kernel<<<NROWS / 8, blk, 0, stream>>>(hbuf, ln1_g + l * DMODEL, ln1_b + l * DMODEL, hn);
    gemm_bf16<EP_BF16><<<gD, blk, 0, stream>>>(hn, wqB + wOff, nullptr, nullptr, nullptr, qB,
                                               NROWS, DMODEL, DMODEL);
    gemm_bf16<EP_KT><<<gD, blk, 0, stream>>>(hn, wkB + wOff, nullptr, nullptr, nullptr, ktB,
                                             NROWS, DMODEL, DMODEL);
    gemm_bf16<EP_BF16><<<gD, blk, 0, stream>>>(hn, wvB + wOff, nullptr, nullptr, nullptr, vB,
                                               NROWS, DMODEL, DMODEL);
    attn_kernel<<<BATCH * NH * (SEQ / 128), blk, 0, stream>>>(qB, ktB, vB, yB);
    gemm_bf16<EP_RESID><<<gD, blk, 0, stream>>>(yB, wpB + wOff, bp + l * DMODEL, hbuf, hbuf,
                                                nullptr, NROWS, DMODEL, DMODEL);
    ln_kernel<<<NROWS / 8, blk, 0, stream>>>(hbuf, ln2_g + l * DMODEL, ln2_b + l * DMODEL, hn);
    gemm_bf16<EP_GELU><<<gF, blk, 0, stream>>>(hn, w1B + (size_t)l * DMODEL * FFD, b1 + l * FFD,
                                               nullptr, nullptr, ffB, NROWS, FFD, DMODEL);
    gemm_bf16<EP_RESID><<<gD, blk, 0, stream>>>(ffB, w2B + (size_t)l * FFD * DMODEL,
                                                b2 + l * DMODEL, hbuf, hbuf, nullptr,
                                                NROWS, DMODEL, FFD);
  }
  ln_kernel<<<NROWS / 8, blk, 0, stream>>>(hbuf, lnf_g, lnf_b, h2B);
  gemm_bf16<EP_OUT><<<gV, blk, 0, stream>>>(h2B, woutB, nullptr, nullptr, (float*)d_out, nullptr,
                                            NROWS, VOC, DMODEL);
}
